// MoELayer_41626823033471
// MI455X (gfx1250) — matrix operand order repaired and checked by reading, not compiled, not run on hardware
//
#include <hip/hip_runtime.h>
#include <hip/hip_bf16.h>
#include <stdint.h>

// ---------------------------------------------------------------------------
// MoE layer for MI455X (gfx1250): gather-based routing + bf16 WMMA expert GEMMs
// - weights pre-converted fp32->bf16 once per launch (no cvt in GEMM loops)
// - A tiles staged to LDS via GLOBAL_LOAD_ASYNC_TO_LDS (ASYNCcnt-tracked DMA)
// - 64x128 block tile, 8 waves, 4 WMMA accumulators per wave, K-chunk 64
// ---------------------------------------------------------------------------

typedef __attribute__((ext_vector_type(16))) __bf16 v16bf;
typedef __attribute__((ext_vector_type(8)))  __bf16 v8bf;
typedef __attribute__((ext_vector_type(4)))  __bf16 v4bf;
typedef __attribute__((ext_vector_type(8)))  float  v8f;

static constexpr int NTOK = 4 * 2048;   // B*S tokens
static constexpr int DM   = 1024;       // model dim
static constexpr int FF   = 4096;       // hidden dim (4D)
static constexpr int NE   = 8;          // experts
static constexpr int CAP  = NTOK;       // worst-case tokens per expert
static constexpr int KC   = 64;         // K-chunk staged in LDS per barrier pair

#if __has_builtin(__builtin_amdgcn_global_load_async_to_lds_b128)
#define HAVE_ASYNC_LDS 1
// Exact element type from the builtin's signature: int vector_size(16).
typedef int v4i __attribute__((vector_size(16)));
typedef __attribute__((address_space(1))) v4i glb_v4i;   // global
typedef __attribute__((address_space(3))) v4i lds_v4i;   // LDS
// flat->LDS aperture maps by low-32-bit truncation (ISA 10.2); AS1 is
// numerically identical to the flat address for global memory.
static __device__ inline lds_v4i* to_lds(void* p) {
  return (lds_v4i*)(unsigned)(uintptr_t)p;
}
static __device__ inline glb_v4i* to_glb(const void* p) {
  return (glb_v4i*)(uintptr_t)p;
}
static __device__ inline void wait_async0() {
#if __has_builtin(__builtin_amdgcn_s_wait_asynccnt)
  __builtin_amdgcn_s_wait_asynccnt(0);
#else
  asm volatile("s_wait_asynccnt 0" ::: "memory");
#endif
}
#endif

// D = A*B + C, bf16 inputs, f32 accumulate (16x16x32)
static __device__ inline v8f wmma_bf16(v16bf a, v16bf b, v8f c) {
  return __builtin_amdgcn_wmma_f32_16x16x32_bf16(
      /*neg_a=*/false, a, /*neg_b=*/false, b,
      /*c_mod=*/(short)0, c, /*reuse_a=*/false, /*reuse_b=*/false);
}

// B fragment: 16 contiguous bf16 K-values of one column (two 16B loads).
static __device__ inline v16bf load_b_frag(const __bf16* p) {
  v8bf lo = *(const v8bf*)p;
  v8bf hi = *(const v8bf*)(p + 8);
  return __builtin_shufflevector(lo, hi, 0, 1, 2, 3, 4, 5, 6, 7,
                                 8, 9, 10, 11, 12, 13, 14, 15);
}

// A fragment (16x32 bf16) from LDS tile [64 rows x KC], row stride KC.
// Contiguous k order, as in B: lanes 0-15 / 16-31 hold the same M rows; lane half selects
// K chunks {0..15} vs {16..31} of the 32-wide step at ks.
static __device__ inline v16bf load_a_frag(const __bf16* As, int mbase, int ks,
                                           int lane) {
  int mr = lane & 15;
  int half = lane >> 4;
  const __bf16* p0 = As + (mbase + mr) * KC + ks + half * 16;
  v8bf lo = *(const v8bf*)p0;
  v8bf hi = *(const v8bf*)(p0 + 8);
  return __builtin_shufflevector(lo, hi, 0, 1, 2, 3, 4, 5, 6, 7,
                                 8, 9, 10, 11, 12, 13, 14, 15);
}

// Stage 32 bytes (16 bf16) of one A row chunk into LDS (per thread).
static __device__ inline void stage_a(__bf16* dst, const __bf16* src) {
#ifdef HAVE_ASYNC_LDS
  __builtin_amdgcn_global_load_async_to_lds_b128(to_glb(src), to_lds(dst), 0, 0);
  __builtin_amdgcn_global_load_async_to_lds_b128(to_glb(src), to_lds(dst), 16, 0);
#else
  *(v8bf*)dst = *(const v8bf*)src;
  *(v8bf*)(dst + 8) = *(const v8bf*)(src + 8);
#endif
}
static __device__ inline void stage_fence() {
#ifdef HAVE_ASYNC_LDS
  wait_async0();
#endif
}

// --------------------------- utility kernels -------------------------------

__global__ void moe_zero(float* __restrict__ out, int* __restrict__ cnt) {
  size_t i = (size_t)blockIdx.x * blockDim.x + threadIdx.x;
  if (i < (size_t)NTOK * DM) out[i] = 0.0f;
  if (i < NE) cnt[i] = 0;
}

// Vectorized fp32 -> bf16 conversion (4 elements / thread).
__global__ void moe_cvt4(const float4* __restrict__ src, v4bf* __restrict__ dst,
                         long n4) {
  long i = (long)blockIdx.x * blockDim.x + threadIdx.x;
  if (i < n4) {
    float4 f = src[i];
    v4bf o;
    o[0] = (__bf16)f.x; o[1] = (__bf16)f.y;
    o[2] = (__bf16)f.z; o[3] = (__bf16)f.w;
    dst[i] = o;
  }
}

// ------------------------------- gating ------------------------------------
__global__ void moe_gate(const float* __restrict__ x, const float* __restrict__ gw,
                         int* __restrict__ cnt, int* __restrict__ toks,
                         float* __restrict__ gates) {
  int wid = (int)((blockIdx.x * blockDim.x + threadIdx.x) >> 5);
  int lane = threadIdx.x & 31;
  if (wid >= NTOK) return;
  const float* xr = x + (size_t)wid * DM;

  float acc[NE] = {};
  for (int k = lane; k < DM; k += 32) {
    float xv = xr[k];
#pragma unroll
    for (int e = 0; e < NE; ++e) acc[e] += xv * gw[e * DM + k];
  }
#pragma unroll
  for (int e = 0; e < NE; ++e)
#pragma unroll
    for (int off = 16; off > 0; off >>= 1)
      acc[e] += __shfl_xor(acc[e], off, 32);

  if (lane == 0) {
    int b0 = 0; float v0 = acc[0];
#pragma unroll
    for (int e = 1; e < NE; ++e)
      if (acc[e] > v0) { v0 = acc[e]; b0 = e; }
    int b1 = (b0 == 0) ? 1 : 0; float v1 = -1e30f;
#pragma unroll
    for (int e = 0; e < NE; ++e)
      if (e != b0 && acc[e] > v1) { v1 = acc[e]; b1 = e; }
    float g0 = 1.0f / (1.0f + __expf(v1 - v0));  // softmax over top-2
    float g1 = 1.0f - g0;
    int s0 = atomicAdd(&cnt[b0], 1);
    toks[b0 * CAP + s0] = wid; gates[b0 * CAP + s0] = g0;
    int s1 = atomicAdd(&cnt[b1], 1);
    toks[b1 * CAP + s1] = wid; gates[b1 * CAP + s1] = g1;
  }
}

// ------------------------- expert GEMM stage 1 -----------------------------
// h = relu(Xg @ W1e^T): M = tokens-of-expert (gathered), N = FF, K = DM.
__global__ void moe_ffn1(const __bf16* __restrict__ xb, const __bf16* __restrict__ w1e,
                         const int* __restrict__ cnt_p, const int* __restrict__ toks,
                         __bf16* __restrict__ h) {
  __shared__ __bf16 As[64 * KC];
  __shared__ int s_tok[64];

  int cnt = *cnt_p;
  int tm = blockIdx.x;
  if (tm * 64 >= cnt) return;
  int tn = blockIdx.y;
  int tid = threadIdx.x;

  if (tid < 64) {
    int m = tm * 64 + tid;
    s_tok[tid] = (m < cnt) ? toks[m] : 0;
  }
  __syncthreads();

  int lane = tid & 31, w = tid >> 5;
  int mb = (w & 1) * 32;               // wave M base within block tile
  int nb = tn * 128 + (w >> 1) * 32;   // wave N base (global)

  v8f c00 = {}, c01 = {}, c10 = {}, c11 = {};
  const __bf16* pB0 = w1e + (size_t)(nb + (lane & 15)) * DM + ((lane >> 4) * 16);
  const __bf16* pB1 = pB0 + (size_t)16 * DM;
  int srow = tid >> 2, sko = (tid & 3) * 16;   // 64 rows x 16 bf16 per thread
  const __bf16* srcA = xb + (size_t)s_tok[srow] * DM + sko;
  __bf16* dstA = As + srow * KC + sko;

  for (int kk = 0; kk < DM; kk += KC) {
    __syncthreads();
    stage_a(dstA, srcA + kk);
    __builtin_prefetch(pB0 + kk + KC, 0, 1);
    __builtin_prefetch(pB1 + kk + KC, 0, 1);
    stage_fence();
    __syncthreads();
#pragma unroll
    for (int ks = 0; ks < KC; ks += 32) {
      v16bf a0 = load_a_frag(As, mb, ks, lane);
      v16bf a1 = load_a_frag(As, mb + 16, ks, lane);
      v16bf b0 = load_b_frag(pB0 + kk + ks);
      v16bf b1 = load_b_frag(pB1 + kk + ks);
      c00 = wmma_bf16(a0, b0, c00);
      c01 = wmma_bf16(a0, b1, c01);
      c10 = wmma_bf16(a1, b0, c10);
      c11 = wmma_bf16(a1, b1, c11);
    }
  }

  // ReLU + store bf16 h; C layout: lane owns (m = r + 8*(lane>>4), n = lane&15)
  int half = lane >> 4, nc = lane & 15;
#pragma unroll
  for (int r = 0; r < 8; ++r) {
    int slot0 = tm * 64 + mb + half * 8 + r;
    int slot1 = slot0 + 16;
    if (slot0 < cnt) {
      h[(size_t)slot0 * FF + nb + nc]      = (__bf16)fmaxf(c00[r], 0.0f);
      h[(size_t)slot0 * FF + nb + 16 + nc] = (__bf16)fmaxf(c01[r], 0.0f);
    }
    if (slot1 < cnt) {
      h[(size_t)slot1 * FF + nb + nc]      = (__bf16)fmaxf(c10[r], 0.0f);
      h[(size_t)slot1 * FF + nb + 16 + nc] = (__bf16)fmaxf(c11[r], 0.0f);
    }
  }
}

// ------------------------- expert GEMM stage 2 -----------------------------
// out[token] += gate * (h @ W2e^T): M = tokens-of-expert, N = DM, K = FF.
// Plain RMW on out is race-free: a token appears at most once per expert and
// expert kernels serialize on the stream.
__global__ void moe_ffn2(const __bf16* __restrict__ h, const __bf16* __restrict__ w2e,
                         const int* __restrict__ cnt_p, const int* __restrict__ toks,
                         const float* __restrict__ gates, float* __restrict__ out) {
  __shared__ __bf16 As[64 * KC];
  __shared__ int s_tok[64];
  __shared__ float s_g[64];

  int cnt = *cnt_p;
  int tm = blockIdx.x;
  if (tm * 64 >= cnt) return;
  int tn = blockIdx.y;
  int tid = threadIdx.x;

  if (tid < 64) {
    int m = tm * 64 + tid;
    s_tok[tid] = (m < cnt) ? toks[m] : 0;
    s_g[tid]   = (m < cnt) ? gates[m] : 0.0f;
  }

  int lane = tid & 31, w = tid >> 5;
  int mb = (w & 1) * 32;
  int nb = tn * 128 + (w >> 1) * 32;

  v8f c00 = {}, c01 = {}, c10 = {}, c11 = {};
  const __bf16* pB0 = w2e + (size_t)(nb + (lane & 15)) * FF + ((lane >> 4) * 16);
  const __bf16* pB1 = pB0 + (size_t)16 * FF;
  int srow = tid >> 2, sko = (tid & 3) * 16;
  int gm = tm * 64 + srow;
  // rows beyond cnt feed only discarded output slots; clamp to row 0 (valid).
  int gsrc = (gm < cnt) ? gm : 0;
  const __bf16* srcA = h + (size_t)gsrc * FF + sko;
  __bf16* dstA = As + srow * KC + sko;

  for (int kk = 0; kk < FF; kk += KC) {
    __syncthreads();
    stage_a(dstA, srcA + kk);
    __builtin_prefetch(pB0 + kk + KC, 0, 1);
    __builtin_prefetch(pB1 + kk + KC, 0, 1);
    stage_fence();
    __syncthreads();
#pragma unroll
    for (int ks = 0; ks < KC; ks += 32) {
      v16bf a0 = load_a_frag(As, mb, ks, lane);
      v16bf a1 = load_a_frag(As, mb + 16, ks, lane);
      v16bf b0 = load_b_frag(pB0 + kk + ks);
      v16bf b1 = load_b_frag(pB1 + kk + ks);
      c00 = wmma_bf16(a0, b0, c00);
      c01 = wmma_bf16(a0, b1, c01);
      c10 = wmma_bf16(a1, b0, c10);
      c11 = wmma_bf16(a1, b1, c11);
    }
  }

  int half = lane >> 4, nc = lane & 15;
#pragma unroll
  for (int r = 0; r < 8; ++r) {
    int local0 = mb + half * 8 + r;
    int local1 = local0 + 16;
    int slot0 = tm * 64 + local0;
    int slot1 = tm * 64 + local1;
    if (slot0 < cnt) {
      float g = s_g[local0];
      float* p = out + (size_t)s_tok[local0] * DM + nb + nc;
      p[0]  += g * c00[r];
      p[16] += g * c01[r];
    }
    if (slot1 < cnt) {
      float g = s_g[local1];
      float* p = out + (size_t)s_tok[local1] * DM + nb + nc;
      p[0]  += g * c10[r];
      p[16] += g * c11[r];
    }
  }
}

// ------------------------------- launcher ----------------------------------

extern "C" void kernel_launch(void* const* d_in, const int* in_sizes, int n_in,
                              void* d_out, int out_size, void* d_ws, size_t ws_size,
                              hipStream_t stream) {
  const float* x  = (const float*)d_in[0];  // [B,S,D]
  const float* gw = (const float*)d_in[1];  // [E,D]
  const float* w1 = (const float*)d_in[2];  // [E,FF,D]
  const float* w2 = (const float*)d_in[3];  // [E,D,FF]
  float* out = (float*)d_out;               // [B,S,D]

  char* ws = (char*)d_ws;
  size_t off = 0;
  __bf16* xb = (__bf16*)(ws + off);    off += (size_t)NTOK * DM * sizeof(__bf16);     // 16 MB
  int* cnt = (int*)(ws + off);         off += 256;
  int* toks = (int*)(ws + off);        off += (size_t)NE * CAP * sizeof(int);          // 256 KB
  float* gates = (float*)(ws + off);   off += (size_t)NE * CAP * sizeof(float);        // 256 KB
  __bf16* h = (__bf16*)(ws + off);     off += (size_t)CAP * FF * sizeof(__bf16);       // 64 MB
  __bf16* w1b = (__bf16*)(ws + off);   off += (size_t)NE * FF * DM * sizeof(__bf16);   // 64 MB
  __bf16* w2b = (__bf16*)(ws + off);   off += (size_t)NE * DM * FF * sizeof(__bf16);   // 64 MB
  (void)ws_size; (void)in_sizes; (void)n_in; (void)out_size;

  const long n_x = (long)NTOK * DM;        // 8.39M
  const long n_w = (long)NE * FF * DM;     // 33.55M per weight tensor

  moe_zero<<<(int)((n_x + 255) / 256), 256, 0, stream>>>(out, cnt);
  moe_cvt4<<<(int)((n_x / 4 + 255) / 256), 256, 0, stream>>>(
      (const float4*)x, (v4bf*)xb, n_x / 4);
  moe_cvt4<<<(int)((n_w / 4 + 255) / 256), 256, 0, stream>>>(
      (const float4*)w1, (v4bf*)w1b, n_w / 4);
  moe_cvt4<<<(int)((n_w / 4 + 255) / 256), 256, 0, stream>>>(
      (const float4*)w2, (v4bf*)w2b, n_w / 4);
  moe_gate<<<(NTOK * 32) / 256, 256, 0, stream>>>(x, gw, cnt, toks, gates);

  for (int e = 0; e < NE; ++e) {
    const __bf16* w1e = w1b + (size_t)e * FF * DM;
    const __bf16* w2e = w2b + (size_t)e * DM * FF;
    moe_ffn1<<<dim3(CAP / 64, FF / 128), 256, 0, stream>>>(xb, w1e, cnt + e,
                                                           toks + e * CAP, h);
    moe_ffn2<<<dim3(CAP / 64, DM / 128), 256, 0, stream>>>(h, w2e, cnt + e,
                                                           toks + e * CAP,
                                                           gates + e * CAP, out);
  }
}
